// MultiHeadAttention_712964571620
// MI455X (gfx1250) — matrix operand order repaired and checked by reading, not compiled, not run on hardware
//
#include <hip/hip_runtime.h>
#include <hip/hip_bf16.h>

// ---------------------------------------------------------------------------
// MHA forward for MI455X (gfx1250): bf16 WMMA (v_wmma_f32_16x16x32_bf16)
// for QKV proj, flash attention (QK^T + PV), and output projection.
// ---------------------------------------------------------------------------

typedef __attribute__((ext_vector_type(16))) __bf16 bf16x16;
typedef __attribute__((ext_vector_type(8)))  __bf16 bf16x8;
typedef __attribute__((ext_vector_type(8)))  float  f32x8;
typedef __attribute__((ext_vector_type(4)))  float  f32x4;

static constexpr int Bc = 4;
static constexpr int Sc = 2048;
static constexpr int Dc = 2048;
static constexpr int Hc = 16;
static constexpr int DH = 128;          // head dim
static constexpr float SCALE = 0.08838834764831845f;  // 1/sqrt(128)

__device__ __forceinline__ f32x8 wmma_bf16(bf16x16 a, bf16x16 b, f32x8 c) {
  return __builtin_amdgcn_wmma_f32_16x16x32_bf16(false, a, false, b,
                                                 (short)0, c, false, false);
}

__device__ __forceinline__ bf16x16 cat16(bf16x8 lo, bf16x8 hi) {
  return __builtin_shufflevector(lo, hi, 0, 1, 2, 3, 4, 5, 6, 7,
                                 8, 9, 10, 11, 12, 13, 14, 15);
}

// Stage 16 contiguous elements (f32 source -> bf16 LDS)
__device__ __forceinline__ void stage16(__bf16* dst, const float* src) {
  f32x4 a0 = ((const f32x4*)src)[0];
  f32x4 a1 = ((const f32x4*)src)[1];
  f32x4 a2 = ((const f32x4*)src)[2];
  f32x4 a3 = ((const f32x4*)src)[3];
  bf16x8 lo, hi;
#pragma unroll
  for (int i = 0; i < 4; ++i) {
    lo[i]     = (__bf16)a0[i];
    lo[i + 4] = (__bf16)a1[i];
    hi[i]     = (__bf16)a2[i];
    hi[i + 4] = (__bf16)a3[i];
  }
  ((bf16x8*)dst)[0] = lo;
  ((bf16x8*)dst)[1] = hi;
}

// Stage 16 contiguous elements (bf16 source -> bf16 LDS)
__device__ __forceinline__ void stage16(__bf16* dst, const __bf16* src) {
  ((bf16x8*)dst)[0] = ((const bf16x8*)src)[0];
  ((bf16x8*)dst)[1] = ((const bf16x8*)src)[1];
}

// ---------------------------------------------------------------------------
// GEMM: C[M,N] = A[M,K] * B[N,K]^T   (einsum 'mk,nk->mn', i.e. x @ W^T)
// A: f32 or bf16 row-major.  B: f32 row-major [N,K].  C: f32 or bf16.
// Block tile 128x128x32, 256 threads = 8 waves (4 along M x 2 along N),
// each wave: 2 M-tiles x 4 N-tiles of 16x16 f32 accumulators.
// ---------------------------------------------------------------------------
template <typename TA, typename TO>
__global__ __launch_bounds__(256) void gemm_wmma(const TA* __restrict__ A,
                                                 const float* __restrict__ B,
                                                 TO* __restrict__ C,
                                                 int M, int N, int K) {
  constexpr int BM = 128, BN = 128, BK = 32, LDT = BK + 8;
  __shared__ __bf16 As[BM * LDT];
  __shared__ __bf16 Bs[BN * LDT];

  const int tid  = threadIdx.x;
  const int lane = tid & 31;
  const int w    = tid >> 5;
  const int wm   = w >> 1;          // 0..3
  const int wn   = w & 1;           // 0..1
  const int l15  = lane & 15;
  const int h16  = lane >> 4;       // 0 or 1
  const int m0   = blockIdx.y * BM;
  const int n0   = blockIdx.x * BN;

  const int srow = tid >> 1;              // 0..127 staged row per thread
  const int sks  = (tid & 1) * 16;        // 0 or 16 within BK

  f32x8 acc[2][4];
#pragma unroll
  for (int i = 0; i < 2; ++i)
#pragma unroll
    for (int j = 0; j < 4; ++j) acc[i][j] = (f32x8)(0.0f);

  for (int k0 = 0; k0 < K; k0 += BK) {
    __syncthreads();
    stage16(&As[srow * LDT + sks], A + (size_t)(m0 + srow) * K + k0 + sks);
    stage16(&Bs[srow * LDT + sks], B + (size_t)(n0 + srow) * K + k0 + sks);
    if (k0 + BK < K) {  // hint next tile into cache (global_prefetch_b8)
      __builtin_prefetch(A + (size_t)(m0 + srow) * K + k0 + BK + sks, 0, 1);
      __builtin_prefetch(B + (size_t)(n0 + srow) * K + k0 + BK + sks, 0, 1);
    }
    __syncthreads();

    bf16x16 af[2], bfr[4];
#pragma unroll
    for (int i = 0; i < 2; ++i) {
      const __bf16* base = &As[(wm * 32 + i * 16 + l15) * LDT];
      bf16x8 lo = *(const bf16x8*)(base + (h16 ? 16 : 0));
      bf16x8 hi = *(const bf16x8*)(base + 8 + (h16 ? 16 : 0));
      af[i] = cat16(lo, hi);
    }
#pragma unroll
    for (int j = 0; j < 4; ++j) {
      const __bf16* base = &Bs[(wn * 64 + j * 16 + l15) * LDT + (h16 ? 16 : 0)];
      bfr[j] = cat16(*(const bf16x8*)base, *(const bf16x8*)(base + 8));
    }
#pragma unroll
    for (int i = 0; i < 2; ++i)
#pragma unroll
      for (int j = 0; j < 4; ++j) acc[i][j] = wmma_bf16(af[i], bfr[j], acc[i][j]);
  }

#pragma unroll
  for (int i = 0; i < 2; ++i)
#pragma unroll
    for (int j = 0; j < 4; ++j)
#pragma unroll
      for (int r = 0; r < 8; ++r) {
        int row = m0 + wm * 32 + i * 16 + r + 8 * h16;
        int col = n0 + wn * 64 + j * 16 + l15;
        C[(size_t)row * N + col] = (TO)acc[i][j][r];
      }
}

// ---------------------------------------------------------------------------
// Flash attention: one block = (b, h, 128-query slab); 8 waves, each owns a
// 16-query tile.  Streams 32 keys/values per step through LDS; online softmax.
// ---------------------------------------------------------------------------
__global__ __launch_bounds__(256) void flash_attn(
    const __bf16* __restrict__ qkv,   // [B,S,3D] (q | k | v)
    const float* __restrict__ bias,   // [H,S]
    const unsigned char* __restrict__ pad,  // [B,S] bool (numpy bool8)
    __bf16* __restrict__ out)         // [B,S,D]
{
  constexpr int LDK = DH + 8;   // K tile row-major  [32][128]
  constexpr int LDV = 32 + 8;   // V tile transposed [128][32]
  constexpr int LDP = 32 + 8;   // P tile            [16][32] per wave
  __shared__ __bf16 Kl[32 * LDK];
  __shared__ __bf16 Vt[DH * LDV];
  __shared__ __bf16 Pl[8 * 16 * LDP];

  const int tid  = threadIdx.x;
  const int lane = tid & 31;
  const int w    = tid >> 5;
  const int l15  = lane & 15;
  const int h16  = lane >> 4;

  const int blk = blockIdx.x;                 // B*H*(S/128) blocks
  const int qb  = blk & 15;
  const int h   = (blk >> 4) & 15;
  const int b   = blk >> 8;
  const int q0  = qb * 128;
  const int q0w = q0 + w * 16;                // wave's first query row
  const int q_hi = q0w + 15;

  // ---- preload Q fragments (16 rows x 128 d, 4 K-chunks of 32) ----
  const int qrow = q0w + l15;
  const __bf16* qbase =
      qkv + ((size_t)b * Sc + qrow) * (3 * Dc) + (size_t)h * DH;
  bf16x16 qf[4];
#pragma unroll
  for (int c = 0; c < 4; ++c) {
    bf16x8 lo = *(const bf16x8*)(qbase + 32 * c + (h16 ? 16 : 0));
    bf16x8 hi = *(const bf16x8*)(qbase + 32 * c + 8 + (h16 ? 16 : 0));
    qf[c] = cat16(lo, hi);
  }

  f32x8 acc[8];
#pragma unroll
  for (int t = 0; t < 8; ++t) acc[t] = (f32x8)(0.0f);
  float mrow[8], lrow[8];
#pragma unroll
  for (int r = 0; r < 8; ++r) { mrow[r] = -3.0e38f; lrow[r] = 0.0f; }

  const int koff = h16 ? 16 : 0;
  const int nj = (q0 + 128) >> 5;             // 32-key blocks needed by slab

  for (int j = 0; j < nj; ++j) {
    __syncthreads();
    {   // ---- stage K (row-major) and V (transposed) into LDS ----
      const int e0 = tid * 16;
      const int kr = e0 >> 7;                 // key row 0..31
      const int cc = e0 & 127;                // d column
      const size_t rowbase = ((size_t)b * Sc + (size_t)j * 32 + kr) * (3 * Dc);
      const __bf16* kp = qkv + rowbase + Dc + (size_t)h * DH + cc;
      stage16(&Kl[kr * LDK + cc], kp);
      const __bf16* vp = qkv + rowbase + 2 * Dc + (size_t)h * DH + cc;
      bf16x8 vlo = ((const bf16x8*)vp)[0];
      bf16x8 vhi = ((const bf16x8*)vp)[1];
#pragma unroll
      for (int i = 0; i < 8; ++i) {
        Vt[(cc + i) * LDV + kr]     = vlo[i];
        Vt[(cc + 8 + i) * LDV + kr] = vhi[i];
      }
    }
    __syncthreads();

    if (j * 32 <= q_hi) {
      // ---- scores: 16 queries x 32 keys = two 16x16 WMMA tiles ----
      f32x8 s0 = (f32x8)(0.0f), s1 = (f32x8)(0.0f);
#pragma unroll
      for (int c = 0; c < 4; ++c) {
        const __bf16* k0p = &Kl[l15 * LDK + 32 * c + koff];
        const __bf16* k1p = &Kl[(16 + l15) * LDK + 32 * c + koff];
        bf16x16 b0 = cat16(*(const bf16x8*)k0p, *(const bf16x8*)(k0p + 8));
        bf16x16 b1 = cat16(*(const bf16x8*)k1p, *(const bf16x8*)(k1p + 8));
        s0 = wmma_bf16(qf[c], b0, s0);
        s1 = wmma_bf16(qf[c], b1, s1);
      }

      const int key0 = j * 32 + l15;
      const int key1 = key0 + 16;
      const float bv0 = bias[h * Sc + key0];
      const float bv1 = bias[h * Sc + key1];
      const bool ok0 = pad[b * Sc + key0] != 0;
      const bool ok1 = pad[b * Sc + key1] != 0;

      float p0e[8], p1e[8];
#pragma unroll
      for (int r = 0; r < 8; ++r) {
        const int qr = q0w + r + 8 * h16;
        float v0 = (ok0 && key0 <= qr) ? s0[r] * SCALE + bv0 : -3.0e38f;
        float v1 = (ok1 && key1 <= qr) ? s1[r] * SCALE + bv1 : -3.0e38f;
        float t = fmaxf(v0, v1);
        t = fmaxf(t, __shfl_xor(t, 1, 32));
        t = fmaxf(t, __shfl_xor(t, 2, 32));
        t = fmaxf(t, __shfl_xor(t, 4, 32));
        t = fmaxf(t, __shfl_xor(t, 8, 32));
        const float mnew = fmaxf(mrow[r], t);
        const float alpha = __expf(mrow[r] - mnew);
        const float e0 = __expf(v0 - mnew);
        const float e1 = __expf(v1 - mnew);
        float srow = e0 + e1;
        srow += __shfl_xor(srow, 1, 32);
        srow += __shfl_xor(srow, 2, 32);
        srow += __shfl_xor(srow, 4, 32);
        srow += __shfl_xor(srow, 8, 32);
        lrow[r] = lrow[r] * alpha + srow;
        mrow[r] = mnew;
#pragma unroll
        for (int t8 = 0; t8 < 8; ++t8) acc[t8][r] *= alpha;
        p0e[r] = e0;
        p1e[r] = e1;
      }

      // ---- re-layout P (C layout -> A layout) via per-wave LDS patch ----
      __bf16* Pw = &Pl[w * 16 * LDP];
#pragma unroll
      for (int r = 0; r < 8; ++r) {
        const int rr = r + 8 * h16;
        Pw[rr * LDP + l15]      = (__bf16)p0e[r];
        Pw[rr * LDP + 16 + l15] = (__bf16)p1e[r];
      }
      // LDS ops are in-order within a wave: safe to read back immediately.
      bf16x16 pf;
      {
        const __bf16* pp = Pw + l15 * LDP;
        bf16x8 lo = *(const bf16x8*)(pp + (h16 ? 16 : 0));
        bf16x8 hi = *(const bf16x8*)(pp + 8 + (h16 ? 16 : 0));
        pf = cat16(lo, hi);
      }

      // ---- PV: acc[16x128] += P[16x32] @ V[32x128] ----
#pragma unroll
      for (int t8 = 0; t8 < 8; ++t8) {
        const __bf16* vp = &Vt[(t8 * 16 + l15) * LDV + koff];
        bf16x16 vf = cat16(*(const bf16x8*)vp, *(const bf16x8*)(vp + 8));
        acc[t8] = wmma_bf16(pf, vf, acc[t8]);
      }
    }
  }

  // ---- normalize and store (bf16) ----
#pragma unroll
  for (int r = 0; r < 8; ++r) {
    const float inv = 1.0f / lrow[r];
    const int row = q0w + r + 8 * h16;
#pragma unroll
    for (int t8 = 0; t8 < 8; ++t8) {
      const int col = h * DH + t8 * 16 + l15;
      out[((size_t)b * Sc + row) * Dc + col] = (__bf16)(acc[t8][r] * inv);
    }
  }
}

// ---------------------------------------------------------------------------
extern "C" void kernel_launch(void* const* d_in, const int* in_sizes, int n_in,
                              void* d_out, int out_size, void* d_ws,
                              size_t ws_size, hipStream_t stream) {
  (void)in_sizes; (void)n_in; (void)out_size; (void)ws_size;
  const float* x     = (const float*)d_in[0];          // [B,S,D]
  const float* Wqkv  = (const float*)d_in[1];          // [3D,D]
  const float* Wout  = (const float*)d_in[2];          // [D,D]
  const float* bias  = (const float*)d_in[3];          // [1,H,1,S]
  const unsigned char* mask = (const unsigned char*)d_in[4];  // [B,S] bool8

  const int M = Bc * Sc;         // 8192 rows
  __bf16* qkv  = (__bf16*)d_ws;                         // [M, 3D] bf16 (96 MB)
  __bf16* attn = qkv + (size_t)M * 3 * Dc;              // [M, D]  bf16 (32 MB)
  float*  out  = (float*)d_out;                         // [M, D]  f32

  // 1) qkv = x @ Wqkv^T   (M x 3D, K = D)
  gemm_wmma<float, __bf16>
      <<<dim3((3 * Dc) / 128, M / 128), 256, 0, stream>>>(x, Wqkv, qkv,
                                                          M, 3 * Dc, Dc);
  // 2) flash attention -> attn (bf16)
  flash_attn<<<dim3(Bc * Hc * (Sc / 128)), 256, 0, stream>>>(qkv, bias, mask,
                                                             attn);
  // 3) out = attn @ Wout^T  (M x D, K = D), f32 result
  gemm_wmma<__bf16, float>
      <<<dim3(Dc / 128, M / 128), 256, 0, stream>>>(attn, Wout, out,
                                                    M, Dc, Dc);
}
